// AP_27711128994522
// MI455X (gfx1250) — matrix operand order repaired and checked by reading, not compiled, not run on hardware
//
#include <hip/hip_runtime.h>
#include <hip/hip_bf16.h>

// ---- problem constants (match reference) ----
#define BATCH   32
#define SLEN    8192
#define DDIM    512
#define NSEED   16
// ---- tiling ----
#define NCHUNK  8                 // s-chunks per batch -> 256 workgroups
#define SCHUNK  (SLEN / NCHUNK)   // 1024
#define SBLOCK  128               // s-block processed per flash step (8 tiles of 16)
#define NBLK    (SCHUNK / SBLOCK) // 8

typedef __attribute__((ext_vector_type(16))) __bf16 v16bf;
typedef __attribute__((ext_vector_type(8)))  float  v8f;

union ABfrag {
    uint4    q[2];
    unsigned u[8];
    v16bf    v;
};

// round-to-nearest-even f32 -> bf16, packed pair into one dword
__device__ __forceinline__ unsigned pack_bf16(float lo, float hi) {
    unsigned a = __float_as_uint(lo);
    unsigned b = __float_as_uint(hi);
    a = (a + 0x7FFFu + ((a >> 16) & 1u)) >> 16;
    b = (b + 0x7FFFu + ((b >> 16) & 1u)) & 0xFFFF0000u;
    return b | (a & 0xFFFFu);
}

// One workgroup = one (batch, s-chunk). 8 waves. Flash-style online softmax.
// x is streamed from HBM exactly once (phase A), staged as bf16 in LDS, and
// re-used from LDS for the P@x GEMM (phase C).
__global__ __launch_bounds__(256)
void ap_attn_pool_partial(const float* __restrict__ x,
                          const float* __restrict__ seeds,
                          float* __restrict__ pooled,   // [256][16][512]
                          float* __restrict__ stats) {  // [256][32] = m[16], l[16]
    __shared__ unsigned sxbf[SBLOCK][DDIM / 2];     // x block as bf16 pairs (128 KB)
    __shared__ unsigned sseed[NSEED][DDIM / 2];     // seeds as bf16 pairs (16 KB)
    __shared__ float    ssc[NSEED][SBLOCK + 4];     // score block (8.25 KB)
    __shared__ unsigned sP[NSEED][SBLOCK / 2];      // exp(weights) bf16 pairs (4 KB)
    __shared__ float    red[NSEED][16];
    __shared__ float    srow_m[NSEED], srow_l[NSEED], sscale[NSEED];

    const int t    = threadIdx.x;
    const int lane = t & 31;
    const int w    = t >> 5;          // wave id 0..7
    const int q    = lane >> 4;       // lane half (WMMA layout selector)
    const int nl   = lane & 15;
    const int b    = blockIdx.x / NCHUNK;
    const int ch   = blockIdx.x % NCHUNK;
    const long s0  = (long)ch * SCHUNK;
    const float* xb = x + (long)b * SLEN * DDIM;

    // stage seeds (f32 -> bf16 pairs) into LDS once
    for (int p = t; p < NSEED * (DDIM / 2); p += 256) {
        int r  = p / (DDIM / 2);
        int c2 = p % (DDIM / 2);
        sseed[r][c2] = pack_bf16(seeds[r * DDIM + 2 * c2],
                                 seeds[r * DDIM + 2 * c2 + 1]);
    }
    if (t < NSEED) { srow_m[t] = -1e30f; srow_l[t] = 0.0f; }
    __syncthreads();

    // pooled accumulators: wave w owns d in [64w, 64w+64) -> 4 16x16 f32 tiles
    v8f acc[4] = {};

    for (int blk = 0; blk < NBLK; ++blk) {
        const long sb = s0 + (long)blk * SBLOCK;

        // ---------- Phase A: scores tile, cols [16w,16w+16) of this block ----
        // Wave w streams x rows [sb+16w, sb+16w+16) from HBM (its lane half
        // covers d = k0+16q..+15), converts to bf16 for the WMMA B operand and
        // stages the packed pairs row-major into LDS for phase C.
        v8f c = {};
        const float* xrow   = xb + (sb + 16 * w + nl) * (long)DDIM + 16 * q;
        const uint4* arow   = (const uint4*)&sseed[nl][0];
        uint4*       xsh_row = (uint4*)&sxbf[16 * w + nl][0];
        for (int k0 = 0; k0 < DDIM; k0 += 32) {
            ABfrag a;
            const int base = (k0 >> 3) + 2 * q;    // uint4 units
            a.q[0] = arow[base];
            a.q[1] = arow[base + 1];
            ABfrag bb;
            const float4* px = (const float4*)(xrow + k0);
            float4 f0 = px[0], f1 = px[1], f2 = px[2], f3 = px[3];
            bb.u[0] = pack_bf16(f0.x, f0.y); bb.u[1] = pack_bf16(f0.z, f0.w);
            bb.u[2] = pack_bf16(f1.x, f1.y); bb.u[3] = pack_bf16(f1.z, f1.w);
            bb.u[4] = pack_bf16(f2.x, f2.y); bb.u[5] = pack_bf16(f2.z, f2.w);
            bb.u[6] = pack_bf16(f3.x, f3.y); bb.u[7] = pack_bf16(f3.z, f3.w);
            xsh_row[(k0 >> 3) + 2 * q]     = bb.q[0];   // stage bf16 x in LDS
            xsh_row[(k0 >> 3) + 2 * q + 1] = bb.q[1];
            c = __builtin_amdgcn_wmma_f32_16x16x32_bf16(
                    false, a.v, false, bb.v, (short)0, c, false, false);
        }
        #pragma unroll
        for (int r = 0; r < 8; ++r)
            ssc[r + 8 * q][16 * w + nl] = c[r];
        __syncthreads();

        // ---------- Phase B: online softmax update over this 128-col block ---
        const int r = t & 15, g = t >> 4;          // 16 threads per row
        float lmax = -1e30f;
        #pragma unroll
        for (int j = 0; j < 8; ++j) lmax = fmaxf(lmax, ssc[r][8 * g + j]);
        red[r][g] = lmax;
        __syncthreads();
        if (t < NSEED) {
            float bm = red[t][0];
            for (int j2 = 1; j2 < 16; ++j2) bm = fmaxf(bm, red[t][j2]);
            float mold = srow_m[t];
            float mnew = fmaxf(mold, bm);
            float sc   = __expf(mold - mnew);
            sscale[t]  = sc;
            srow_l[t] *= sc;
            srow_m[t]  = mnew;
        }
        __syncthreads();
        {
            float mrow = srow_m[r];
            float lsum = 0.0f;
            uint4 pv;
            float p0, p1;
            p0 = __expf(ssc[r][8 * g + 0] - mrow); p1 = __expf(ssc[r][8 * g + 1] - mrow);
            lsum += p0 + p1; pv.x = pack_bf16(p0, p1);
            p0 = __expf(ssc[r][8 * g + 2] - mrow); p1 = __expf(ssc[r][8 * g + 3] - mrow);
            lsum += p0 + p1; pv.y = pack_bf16(p0, p1);
            p0 = __expf(ssc[r][8 * g + 4] - mrow); p1 = __expf(ssc[r][8 * g + 5] - mrow);
            lsum += p0 + p1; pv.z = pack_bf16(p0, p1);
            p0 = __expf(ssc[r][8 * g + 6] - mrow); p1 = __expf(ssc[r][8 * g + 7] - mrow);
            lsum += p0 + p1; pv.w = pack_bf16(p0, p1);
            *((uint4*)&sP[r][4 * g]) = pv;
            red[r][g] = lsum;
        }
        __syncthreads();
        if (t < NSEED) {
            float s = 0.0f;
            for (int j2 = 0; j2 < 16; ++j2) s += red[t][j2];
            srow_l[t] += s;
        }
        __syncthreads();

        // ---------- Phase C: rescale accumulators, pooled += P @ x_block -----
        float scl[8];
        #pragma unroll
        for (int v = 0; v < 8; ++v) scl[v] = sscale[v + 8 * q];
        #pragma unroll
        for (int tt = 0; tt < 4; ++tt)
            #pragma unroll
            for (int v = 0; v < 8; ++v) acc[tt][v] *= scl[v];

        // prefetch next s-block's rows for phase A (global_prefetch_b8)
        if (blk + 1 < NBLK) {
            const char* pf =
                (const char*)(xb + (sb + SBLOCK + 16 * w + nl) * (long)DDIM);
            #pragma unroll
            for (int i = 0; i < 4; ++i)
                __builtin_prefetch(pf + (4 * q + i) * 256, 0, 3);
        }

        const uint4* prow = (const uint4*)&sP[nl][0];
        const unsigned short* xsh = (const unsigned short*)&sxbf[0][0];
        for (int k0 = 0; k0 < SBLOCK; k0 += 32) {
            ABfrag a;
            const int base = (k0 >> 3) + 2 * q;
            a.q[0] = prow[base];
            a.q[1] = prow[base + 1];
            const int krow = k0 + 16 * q;
            #pragma unroll
            for (int tt = 0; tt < 4; ++tt) {
                const int d = 64 * w + 16 * tt + nl;
                ABfrag bb;
                #pragma unroll
                for (int j = 0; j < 8; ++j) {
                    unsigned lo = xsh[(krow + 2 * j) * DDIM + d];
                    unsigned hi = xsh[(krow + 2 * j + 1) * DDIM + d];
                    bb.u[j] = lo | (hi << 16);
                }
                acc[tt] = __builtin_amdgcn_wmma_f32_16x16x32_bf16(
                              false, a.v, false, bb.v, (short)0, acc[tt], false, false);
            }
        }
        __syncthreads();
    }

    // ---------- epilogue: partial pooled + stats -----------------------------
    const long pbase = (long)blockIdx.x * NSEED * DDIM;
    #pragma unroll
    for (int tt = 0; tt < 4; ++tt) {
        const int d = 64 * w + 16 * tt + nl;
        #pragma unroll
        for (int v = 0; v < 8; ++v) {
            const int m = v + 8 * q;
            pooled[pbase + (long)m * DDIM + d] = acc[tt][v];
        }
    }
    if (t < NSEED) {
        stats[blockIdx.x * 32 + t]      = srow_m[t];
        stats[blockIdx.x * 32 + 16 + t] = srow_l[t];
    }
}

// Merge NCHUNK partials per batch: out = sum_c e^{m_c-M} pooled_c / sum_c e^{m_c-M} l_c
__global__ __launch_bounds__(256)
void ap_attn_pool_combine(const float* __restrict__ pooled,
                          const float* __restrict__ stats,
                          float* __restrict__ out) {
    __shared__ float Wc[NCHUNK][NSEED];
    __shared__ float inv_den[NSEED];
    const int b = blockIdx.x;
    const int t = threadIdx.x;
    if (t < NSEED) {
        float mmax = -1e30f;
        for (int c = 0; c < NCHUNK; ++c)
            mmax = fmaxf(mmax, stats[(b * NCHUNK + c) * 32 + t]);
        float den = 0.0f;
        for (int c = 0; c < NCHUNK; ++c) {
            float wgt = __expf(stats[(b * NCHUNK + c) * 32 + t] - mmax);
            Wc[c][t] = wgt;
            den += wgt * stats[(b * NCHUNK + c) * 32 + 16 + t];
        }
        inv_den[t] = 1.0f / den;
    }
    __syncthreads();
    for (int i = t; i < NSEED * DDIM; i += 256) {
        const int m = i >> 9;        // i / DDIM
        const int d = i & (DDIM - 1);
        float acc = 0.0f;
        for (int c = 0; c < NCHUNK; ++c)
            acc += Wc[c][m] * pooled[((long)(b * NCHUNK + c) * NSEED + m) * DDIM + d];
        out[((long)b * NSEED + m) * DDIM + d] = acc * inv_den[m];
    }
}

extern "C" void kernel_launch(void* const* d_in, const int* in_sizes, int n_in,
                              void* d_out, int out_size, void* d_ws, size_t ws_size,
                              hipStream_t stream) {
    const float* x     = (const float*)d_in[0];   // [32, 8192, 512] f32
    const float* seeds = (const float*)d_in[1];   // [16, 512] f32
    float* out    = (float*)d_out;                // [32, 16, 512] f32
    float* pooled = (float*)d_ws;                 // 256*16*512 f32 = 8 MiB
    float* stats  = pooled + (long)BATCH * NCHUNK * NSEED * DDIM; // 256*32 f32

    ap_attn_pool_partial<<<BATCH * NCHUNK, 256, 0, stream>>>(x, seeds, pooled, stats);
    ap_attn_pool_combine<<<BATCH, 256, 0, stream>>>(pooled, stats, out);
}
